// MultiHeadAttention_14482629722463
// MI455X (gfx1250) — matrix operand order repaired and checked by reading, not compiled, not run on hardware
//
#include <hip/hip_runtime.h>

// ---------------------------------------------------------------------------
// MultiHeadAttention for MI455X (gfx1250), wave32 + WMMA f16 (f32 accum).
//   B=2, N=2048, HIDDEN=1024, HEADS=16, DK=64
// ---------------------------------------------------------------------------

#define HIDDEN 1024
#define HEADS  16
#define DKH    64
#define BATCH  2
#define SEQ    2048
#define MTOT   (BATCH * SEQ)   // 4096 rows for the projection GEMMs

typedef __attribute__((ext_vector_type(16))) _Float16 v16h;
typedef __attribute__((ext_vector_type(8)))  _Float16 v8h;
typedef __attribute__((ext_vector_type(8)))  float    v8f;

static __device__ inline v8f v8f_zero() {
  v8f z;
#pragma unroll
  for (int i = 0; i < 8; ++i) z[i] = 0.0f;
  return z;
}

// Build a 16-half A/B fragment from two contiguous 8-half chunks.
static __device__ inline v16h cat8(v8h a, v8h b) {
  return __builtin_shufflevector(a, b, 0, 1, 2, 3, 4, 5, 6, 7,
                                        8, 9, 10, 11, 12, 13, 14, 15);
}

// ---------------------------------------------------------------------------
// Elementwise f32 -> f16 convert.
// ---------------------------------------------------------------------------
__global__ __launch_bounds__(256)
void cvt_f32_f16(const float* __restrict__ s, _Float16* __restrict__ d, int n) {
  int i = blockIdx.x * 256 + threadIdx.x;
  if (i < n) d[i] = (_Float16)s[i];
}

// W[in=1024][out=1024] f32  ->  Wt[out][in] f16   (so B-fragments are
// contiguous 16-element runs of one output channel's column).
__global__ __launch_bounds__(256)
void cvt_w_transpose(const float* __restrict__ W, _Float16* __restrict__ Wt) {
  int i = blockIdx.x * 256 + threadIdx.x;   // 0 .. HIDDEN*HIDDEN-1
  int r = i >> 10;                          // input channel
  int c = i & (HIDDEN - 1);                 // output channel
  Wt[(size_t)c * HIDDEN + r] = (_Float16)W[i];
}

// ---------------------------------------------------------------------------
// WMMA GEMM:  C[M=4096, N=1024] = A[M,K=1024] * Bt^T + bias, then store per
// MODE.  Block = 256 threads = 8 waves; wave tile = 16(M) x 64(N); workgroup
// tile = 64 x 128.  K loop in steps of 32 with v_wmma_f32_16x16x32_f16.
//   MODE 0: f16 store, head-split [B,H,N,DK]   (used for Q with scale=1/8, K)
//   MODE 1: f16 store, transposed  [B,H,DK,N]  (used for V)
//   MODE 2: f32 store, plain [M, HIDDEN]       (final output projection)
// ---------------------------------------------------------------------------
template <int MODE>
__global__ __launch_bounds__(256)
void gemm_wmma(const _Float16* __restrict__ A,
               const _Float16* __restrict__ Bt,   // [N][K] = B^T, row-major
               const float* __restrict__ bias,    // [N]
               void* __restrict__ outp,
               float scale) {
  const int K    = HIDDEN;
  const int lane = threadIdx.x & 31;
  const int wave = threadIdx.x >> 5;
  const int waveM = wave & 3;            // 4 M-slots
  const int waveN = wave >> 2;           // 2 N-slots
  const int mBase = blockIdx.x * 64 + waveM * 16;
  const int nBase = blockIdx.y * 128 + waveN * 64;

  const int col = lane & 15;             // N / M index within a 16-wide tile
  const int hi  = lane >> 4;             // half-wave select
  const int kb0 = hi * 16;               // A-fragment K sub-offset

  v8f acc[4];
#pragma unroll
  for (int t = 0; t < 4; ++t) acc[t] = v8f_zero();

  const _Float16* Arow = A + (size_t)(mBase + col) * K;

  for (int k0 = 0; k0 < K; k0 += 32) {
    if (k0 + 32 < K) {
      __builtin_prefetch(Arow + k0 + 32, 0, 1);          // global_prefetch_b8
    }
    // A fragment: 16x32 f16.  a[0..7]=A[row][k0+kb0..], a[8..15]=+8.
    v8h a0 = *(const v8h*)(Arow + k0 + kb0);
    v8h a1 = *(const v8h*)(Arow + k0 + kb0 + 8);
    v16h afrag = cat8(a0, a1);
#pragma unroll
    for (int t = 0; t < 4; ++t) {
      // B fragment: lane holds 16 contiguous K values of column nBase+t*16+col
      const _Float16* Brow =
          Bt + (size_t)(nBase + t * 16 + col) * K + k0 + hi * 16;
      v16h bfrag = *(const v16h*)Brow;
      acc[t] = __builtin_amdgcn_wmma_f32_16x16x32_f16(
          false, afrag, false, bfrag, (short)0, acc[t], false, false);
    }
  }

  // Epilogue: C VGPR v -> row mBase + v + hi*8, col nBase + t*16 + col.
#pragma unroll
  for (int t = 0; t < 4; ++t) {
    const int n  = nBase + t * 16 + col;
    const float bn = bias[n];
#pragma unroll
    for (int vv = 0; vv < 8; ++vv) {
      const int m  = mBase + vv + hi * 8;
      float val = (acc[t][vv] + bn) * scale;
      if (MODE == 2) {
        ((float*)outp)[(size_t)m * HIDDEN + n] = val;
      } else {
        const int b = m >> 11;               // SEQ = 2048
        const int ns = m & (SEQ - 1);
        const int h = n >> 6;                // DK = 64
        const int dk = n & (DKH - 1);
        size_t idx;
        if (MODE == 1)
          idx = (((size_t)(b * HEADS + h)) * DKH + dk) * SEQ + ns;   // [B,H,DK,N]
        else
          idx = (((size_t)(b * HEADS + h)) * SEQ + ns) * DKH + dk;   // [B,H,N,DK]
        ((_Float16*)outp)[idx] = (_Float16)val;
      }
    }
  }
}

// ---------------------------------------------------------------------------
// Flash attention.  Grid: (SEQ/128, BATCH*HEADS), 256 threads = 8 waves.
// Each wave owns a 16-query tile, loops over keys in blocks of 32.
//   scores 16x32 : 4 WMMAs (dk=64 -> 2 K-steps per 16-key half)
//   probs * V    : 4 WMMAs (16x32 A from LDS, Vt B frags, 64 dk cols)
// Online softmax with 16-lane butterfly reductions (wave32).
// ---------------------------------------------------------------------------
__global__ __launch_bounds__(256)
void attn_wmma(const _Float16* __restrict__ qh,   // [B,H,N,DK], pre-scaled
               const _Float16* __restrict__ kh,   // [B,H,N,DK]
               const _Float16* __restrict__ vt,   // [B,H,DK,N]
               const float* __restrict__ bias,    // [B,N,N]
               const float* __restrict__ mask,    // [B,N,N]
               _Float16* __restrict__ ctx) {      // [B,N,HIDDEN]
  __shared__ _Float16 lds[8][16 * 32];            // per-wave P tile (8 KB)

  const int lane = threadIdx.x & 31;
  const int wave = threadIdx.x >> 5;
  const int col  = lane & 15;
  const int hi   = lane >> 4;
  const int kb0  = hi * 16;

  const int bh = blockIdx.y;                       // 0 .. B*H-1
  const int b  = bh / HEADS;
  const int h  = bh % HEADS;
  const int qBase = (blockIdx.x * 8 + wave) * 16;  // first query row

  const _Float16* qbase = qh + ((size_t)bh * SEQ + qBase) * DKH;
  const _Float16* kbase = kh + (size_t)bh * SEQ * DKH;
  const _Float16* vbase = vt + (size_t)bh * DKH * SEQ;
  const float* biasbase = bias + (size_t)b * SEQ * SEQ;
  const float* maskbase = mask + (size_t)b * SEQ * SEQ;

  // Preload the two Q A-fragments (dk 0..31, 32..63); reused for all keys.
  v16h aq[2];
#pragma unroll
  for (int s = 0; s < 2; ++s) {
    const _Float16* qr = qbase + (size_t)col * DKH + s * 32;
    aq[s] = cat8(*(const v8h*)(qr + kb0), *(const v8h*)(qr + kb0 + 8));
  }

  v8f O[4];
#pragma unroll
  for (int t = 0; t < 4; ++t) O[t] = v8f_zero();
  float rmax[8], rsum[8];
#pragma unroll
  for (int vv = 0; vv < 8; ++vv) { rmax[vv] = -1e30f; rsum[vv] = 0.0f; }

  for (int kb = 0; kb < SEQ; kb += 32) {
    // ---- scores: S[t] = Q(16x64) * K^T(64x16) for key halves t=0,1 ----
    v8f S[2];
    S[0] = v8f_zero(); S[1] = v8f_zero();
#pragma unroll
    for (int t = 0; t < 2; ++t) {
#pragma unroll
      for (int s = 0; s < 2; ++s) {
        const _Float16* kr =
            kbase + (size_t)(kb + t * 16 + col) * DKH + s * 32 + hi * 16;
        v16h bfrag = *(const v16h*)kr;
        S[t] = __builtin_amdgcn_wmma_f32_16x16x32_f16(
            false, aq[s], false, bfrag, (short)0, S[t], false, false);
      }
    }

    // ---- bias + mask, online softmax update, spill P to LDS ----
#pragma unroll
    for (int vv = 0; vv < 8; ++vv) {
      const int m = qBase + vv + hi * 8;           // query row
      const float* bp = biasbase + (size_t)m * SEQ + kb;
      const float* mp = maskbase + (size_t)m * SEQ + kb;
      float s0 = S[0][vv] + bp[col]      + mp[col];
      float s1 = S[1][vv] + bp[16 + col] + mp[16 + col];

      float mx = fmaxf(s0, s1);
#pragma unroll
      for (int off = 1; off <= 8; off <<= 1)       // 16-lane butterfly
        mx = fmaxf(mx, __shfl_xor(mx, off, 32));

      const float mnew  = fmaxf(rmax[vv], mx);
      const float alpha = __expf(rmax[vv] - mnew);
      const float p0 = __expf(s0 - mnew);
      const float p1 = __expf(s1 - mnew);

      float ps = p0 + p1;
#pragma unroll
      for (int off = 1; off <= 8; off <<= 1)
        ps += __shfl_xor(ps, off, 32);

      rsum[vv] = rsum[vv] * alpha + ps;
      rmax[vv] = mnew;
#pragma unroll
      for (int t = 0; t < 4; ++t) O[t][vv] *= alpha;

      const int r = vv + hi * 8;                   // local row 0..15
      lds[wave][r * 32 + col]      = (_Float16)p0;
      lds[wave][r * 32 + 16 + col] = (_Float16)p1;
    }

    // ---- re-layout P as A fragment (wave-private LDS, DS ops in order) ----
    const _Float16* lp = &lds[wave][col * 32 + kb0];
    v16h ap = cat8(*(const v8h*)lp, *(const v8h*)(lp + 8));

    // ---- O(16x64) += P(16x32) * V(32x64) ----
#pragma unroll
    for (int t = 0; t < 4; ++t) {
      const _Float16* vr =
          vbase + (size_t)(t * 16 + col) * SEQ + kb + hi * 16;
      v16h bfrag = *(const v16h*)vr;
      O[t] = __builtin_amdgcn_wmma_f32_16x16x32_f16(
          false, ap, false, bfrag, (short)0, O[t], false, false);
    }
  }

  // ---- normalize and store ctx[b][n][h*64+dk] as f16 ----
#pragma unroll
  for (int t = 0; t < 4; ++t) {
#pragma unroll
    for (int vv = 0; vv < 8; ++vv) {
      const int m  = qBase + vv + hi * 8;
      const int dk = t * 16 + col;
      const float val = O[t][vv] / rsum[vv];
      ctx[((size_t)b * SEQ + m) * HIDDEN + h * DKH + dk] = (_Float16)val;
    }
  }
}

// ---------------------------------------------------------------------------
// Host-side launcher.
// ---------------------------------------------------------------------------
extern "C" void kernel_launch(void* const* d_in, const int* in_sizes, int n_in,
                              void* d_out, int out_size, void* d_ws,
                              size_t ws_size, hipStream_t stream) {
  (void)in_sizes; (void)n_in; (void)out_size; (void)ws_size;
  const float* q  = (const float*)d_in[0];
  const float* k  = (const float*)d_in[1];
  const float* v  = (const float*)d_in[2];
  const float* attn_bias = (const float*)d_in[3];
  const float* attn_mask = (const float*)d_in[4];
  const float* Wq = (const float*)d_in[5];
  const float* bq = (const float*)d_in[6];
  const float* Wk = (const float*)d_in[7];
  const float* bk = (const float*)d_in[8];
  const float* Wv = (const float*)d_in[9];
  const float* bv = (const float*)d_in[10];
  const float* Wo = (const float*)d_in[11];
  const float* bo = (const float*)d_in[12];

  const size_t SZX = (size_t)MTOT * HIDDEN;     // 4,194,304 elems
  const size_t SZW = (size_t)HIDDEN * HIDDEN;   // 1,048,576 elems

  _Float16* xq  = (_Float16*)d_ws;   // f16 activations + intermediates
  _Float16* xk  = xq  + SZX;
  _Float16* xv  = xk  + SZX;
  _Float16* qhB = xv  + SZX;         // [B,H,N,DK], pre-scaled by 1/8
  _Float16* khB = qhB + SZX;         // [B,H,N,DK]
  _Float16* vtB = khB + SZX;         // [B,H,DK,N]
  _Float16* ctx = vtB + SZX;         // [B,N,HIDDEN]
  _Float16* wqt = ctx + SZX;         // transposed f16 weights [out][in]
  _Float16* wkt = wqt + SZW;
  _Float16* wvt = wkt + SZW;
  _Float16* wot = wvt + SZW;

  // 1) convert activations and weights to f16 (weights transposed)
  cvt_f32_f16<<<dim3((unsigned)(SZX / 256)), dim3(256), 0, stream>>>(q, xq, (int)SZX);
  cvt_f32_f16<<<dim3((unsigned)(SZX / 256)), dim3(256), 0, stream>>>(k, xk, (int)SZX);
  cvt_f32_f16<<<dim3((unsigned)(SZX / 256)), dim3(256), 0, stream>>>(v, xv, (int)SZX);
  cvt_w_transpose<<<dim3((unsigned)(SZW / 256)), dim3(256), 0, stream>>>(Wq, wqt);
  cvt_w_transpose<<<dim3((unsigned)(SZW / 256)), dim3(256), 0, stream>>>(Wk, wkt);
  cvt_w_transpose<<<dim3((unsigned)(SZW / 256)), dim3(256), 0, stream>>>(Wv, wvt);
  cvt_w_transpose<<<dim3((unsigned)(SZW / 256)), dim3(256), 0, stream>>>(Wo, wot);

  // 2) projections (WMMA GEMMs); Q pre-scaled by DK^-0.5 = 0.125
  dim3 gg(MTOT / 64, HIDDEN / 128);
  gemm_wmma<0><<<gg, dim3(256), 0, stream>>>(xq, wqt, bq, qhB, 0.125f);
  gemm_wmma<0><<<gg, dim3(256), 0, stream>>>(xk, wkt, bk, khB, 1.0f);
  gemm_wmma<1><<<gg, dim3(256), 0, stream>>>(xv, wvt, bv, vtB, 1.0f);

  // 3) flash attention
  attn_wmma<<<dim3(SEQ / 128, BATCH * HEADS), dim3(256), 0, stream>>>(
      qhB, khB, vtB, attn_bias, attn_mask, ctx);

  // 4) output projection -> f32 d_out
  gemm_wmma<2><<<gg, dim3(256), 0, stream>>>(ctx, wot, bo, d_out, 1.0f);
}
